// Model_45079976739081
// MI455X (gfx1250) — matrix operand order repaired and checked by reading, not compiled, not run on hardware
//
#include <hip/hip_runtime.h>
#include <math.h>

// Problem constants (match reference).
constexpr int kB  = 32;    // batch
constexpr int kL  = 512;   // seq_len
constexpr int kP  = 192;   // pred_len
constexpr int kC  = 512;   // d_model
constexpr int kT  = 97;    // rfft bins of P
constexpr int kTP = 128;   // padded T (multiple of 32 for WMMA K loop)

typedef __attribute__((ext_vector_type(16))) unsigned short v16u;
typedef __attribute__((ext_vector_type(8)))  unsigned short v8u;
typedef __attribute__((ext_vector_type(4)))  unsigned short v4u;
typedef __attribute__((ext_vector_type(16))) __bf16         v16bf;
typedef __attribute__((ext_vector_type(8)))  float          v8f;
typedef __attribute__((ext_vector_type(4)))  float          v4f;

// ---------- bf16 helpers (raw ushort storage, RNE convert) ----------
__device__ inline unsigned short f2bf(float f) {
  unsigned u = __builtin_bit_cast(unsigned, f);
  u += 0x7FFFu + ((u >> 16) & 1u);
  return (unsigned short)(u >> 16);
}

__device__ inline v8f wmma_bf16(v16u a, v16u b, v8f c) {
  // D = A(16x32 bf16) x B(32x16 bf16) + C(16x16 f32)
  return __builtin_amdgcn_wmma_f32_16x16x32_bf16(
      false, __builtin_bit_cast(v16bf, a),
      false, __builtin_bit_cast(v16bf, b),
      (short)0, c, false, false);
}

__device__ inline v16u negbf(v16u a) {
  v16u s;
#pragma unroll
  for (int e = 0; e < 16; ++e) s[e] = 0x8000;
  return a ^ s;                      // bf16 sign flip (exact)
}

// A fragment: element (m,k) at A[m*lda + k] (row-major M x K).
// lane<16 -> M=lane, K = k0 + {0..7, 16..23}; lane>=16 -> M=lane-16, K += 8.
// Two 16-byte loads (requires lda multiple of 8 elements; k0 multiple of 32).
__device__ inline v16u load_frag_a(const unsigned short* __restrict__ A,
                                   int m0, int k0, size_t lda) {
  int lane = threadIdx.x & 31;
  int m  = m0 + (lane & 15);
  int kb = k0 + ((lane & 16) ? 8 : 0);
  const unsigned short* p = A + (size_t)m * lda + kb;
  v8u lo = *(const v8u*)p;          // K = kb + 0..7
  v8u hi = *(const v8u*)(p + 16);   // K = kb + 16..23
  return __builtin_shufflevector(lo, hi, 0, 1, 2, 3, 4, 5, 6, 7,
                                 8, 9, 10, 11, 12, 13, 14, 15);
}

// B fragment from K-contiguous ("transposed") storage: element (k,n) at B[n*ldb + k].
// lane<16 -> N=lane, K = k0 + {0..7, 16..23}; lane>=16 -> N=lane-16, K += 8.
__device__ inline v16u load_frag_bt(const unsigned short* __restrict__ Bm,
                                    int k0, int n0, size_t ldb) {
  int lane = threadIdx.x & 31;
  int n  = n0 + (lane & 15);
  int kb = k0 + ((lane & 16) ? 8 : 0);
  const unsigned short* p = Bm + (size_t)n * ldb + kb;
  v8u lo = *(const v8u*)p;
  v8u hi = *(const v8u*)(p + 16);
  return __builtin_shufflevector(lo, hi, 0, 1, 2, 3, 4, 5, 6, 7,
                                 8, 9, 10, 11, 12, 13, 14, 15);
}

// ---------- Stage 0: RevIN stats ----------
__global__ void stats_kernel(const float* __restrict__ x,
                             float* __restrict__ mean,
                             float* __restrict__ stdev,
                             float* __restrict__ invstd) {
  int b = blockIdx.x;
  for (int c = threadIdx.x; c < kC; c += 256) {
    const float* p = x + (size_t)b * kL * kC + c;
    float s = 0.f, s2 = 0.f;
    for (int l = 0; l < kL; ++l) {
      float v = p[(size_t)l * kC];
      s += v; s2 += v * v;
    }
    float m   = s * (1.0f / kL);
    float var = s2 * (1.0f / kL) - m * m;
    float sd  = sqrtf(var + 1e-5f);
    mean[b * kC + c]   = m;
    stdev[b * kC + c]  = sd;
    invstd[b * kC + c] = 1.0f / sd;
  }
}

// ---------- Stage 0b: normalize + transpose to (b, c, l) bf16 via LDS ----------
__global__ __launch_bounds__(256) void normalize_t_kernel(
    const float* __restrict__ x, const float* __restrict__ mean,
    const float* __restrict__ invstd, unsigned short* __restrict__ xnt) {
  __shared__ unsigned short tile[32][33];
  int b  = blockIdx.z;
  int l0 = blockIdx.y * 32;
  int c0 = blockIdx.x * 32;
  int tx = threadIdx.x & 31;
  int ty = threadIdx.x >> 5;
  for (int i = ty; i < 32; i += 8) {
    int l = l0 + i, c = c0 + tx;
    float v = (x[((size_t)b * kL + l) * kC + c] - mean[b * kC + c]) * invstd[b * kC + c];
    tile[i][tx] = f2bf(v);
  }
  __syncthreads();
  for (int i = ty; i < 32; i += 8) {
    int c = c0 + i, l = l0 + tx;
    xnt[((size_t)b * kC + c) * kL + l] = tile[tx][i];
  }
}

// ---------- Generic f32 -> bf16 convert (vectorized x4) ----------
__global__ void cvt_bf16_kernel(const float* __restrict__ src,
                                unsigned short* __restrict__ dst, size_t n4) {
  size_t stride = (size_t)gridDim.x * blockDim.x;
  for (size_t i = (size_t)blockIdx.x * blockDim.x + threadIdx.x; i < n4; i += stride) {
    v4f v = *(const v4f*)(src + 4 * i);
    v4u o;
#pragma unroll
    for (int j = 0; j < 4; ++j) o[j] = f2bf(v[j]);
    *(v4u*)(dst + 4 * i) = o;
  }
}

// ---------- Fourier basis matrices (bf16), zero-padded rows k >= T ----------
__global__ void basis_kernel(unsigned short* __restrict__ Fc,
                             unsigned short* __restrict__ Fs,
                             unsigned short* __restrict__ Cb,
                             unsigned short* __restrict__ Sb) {
  int idx = blockIdx.x * blockDim.x + threadIdx.x;
  if (idx >= kTP * kP) return;
  int k = idx / kP, l = idx % kP;
  float ang = 6.28318530717958647692f * (float)((k * l) % kP) / (float)kP;
  float cv = __cosf(ang), sv = __sinf(ang);
  bool valid = (k < kT);
  float isq = rsqrtf((float)kP);
  Fc[(size_t)k * kP + l]  = f2bf(valid ?  cv * isq : 0.f);  // rfft real basis (ortho)
  Fs[(size_t)k * kP + l]  = f2bf(valid ? -sv * isq : 0.f);  // rfft imag basis (ortho)
  Cb[(size_t)l * kTP + k] = f2bf(valid ?  cv : 0.f);        // irfft cos basis
  Sb[(size_t)l * kTP + k] = f2bf(valid ? -sv : 0.f);        // irfft (-sin) basis
}

// ---------- softmax weights (two layers), amp folded in ----------
__global__ void wk_kernel(const float* __restrict__ w,
                          float* __restrict__ wk1, float* __restrict__ wk2) {
  __shared__ float sh[128];
  int t = threadIdx.x;
  float v = (t < kT) ? w[t] : -1e30f;
  sh[t] = v; __syncthreads();
  for (int s = 64; s > 0; s >>= 1) { if (t < s) sh[t] = fmaxf(sh[t], sh[t + s]); __syncthreads(); }
  float mx = sh[0]; __syncthreads();
  float e = (t < kT) ? __expf(v - mx) : 0.f;
  sh[t] = e; __syncthreads();
  for (int s = 64; s > 0; s >>= 1) { if (t < s) sh[t] += sh[t + s]; __syncthreads(); }
  float w1 = e / sh[0]; __syncthreads();

  float v2 = (t < kT) ? w1 : -1e30f;
  sh[t] = v2; __syncthreads();
  for (int s = 64; s > 0; s >>= 1) { if (t < s) sh[t] = fmaxf(sh[t], sh[t + s]); __syncthreads(); }
  float mx2 = sh[0]; __syncthreads();
  float e2 = (t < kT) ? __expf(v2 - mx2) : 0.f;
  sh[t] = e2; __syncthreads();
  for (int s = 64; s > 0; s >>= 1) { if (t < s) sh[t] += sh[t + s]; __syncthreads(); }
  float w2 = e2 / sh[0];

  if (t < kT) {
    float amp = ((t > 0) && (t < kT - 1)) ? 2.f : 1.f;
    amp *= rsqrtf((float)kP);
    wk1[t] = w1 * amp;
    wk2[t] = w2 * amp;
  }
}

// ---------- Stage A: predict_linear  X(b) = Wp(PxL) x Xn(b)(LxC) + bp ----------
__global__ __launch_bounds__(256) void gemm_predict(
    const unsigned short* __restrict__ Wp_bf, const unsigned short* __restrict__ XnT,
    const float* __restrict__ bp, float* __restrict__ Xc, unsigned short* __restrict__ XcT) {
  int b = blockIdx.z;
  int tile = blockIdx.x * 8 + (threadIdx.x >> 5);   // 384 tiles = 12 Mt x 32 Nt
  int mt = tile >> 5, nt = tile & 31;
  int m0 = mt * 16, n0 = nt * 16;
  const unsigned short* Bm = XnT + (size_t)b * kC * kL;
  v8f acc = {};
  for (int k0 = 0; k0 < kL; k0 += 32) {
    v16u a  = load_frag_a(Wp_bf, m0, k0, kL);
    v16u bb = load_frag_bt(Bm, k0, n0, kL);
    acc = wmma_bf16(a, bb, acc);
  }
  int lane = threadIdx.x & 31;
  int n = n0 + (lane & 15);
  int mb = m0 + ((lane & 16) ? 8 : 0);
  float* out = Xc + (size_t)b * kP * kC;
  v8u pk;
#pragma unroll
  for (int r = 0; r < 8; ++r) {
    float v = acc[r] + bp[mb + r];
    out[(size_t)(mb + r) * kC + n] = v;
    pk[r] = f2bf(v);
  }
  // transposed bf16 copy (b, c, p): 8 contiguous M values -> one 16B store
  *(v8u*)(XcT + ((size_t)b * kC + n) * kP + mb) = pk;
}

// ---------- Stage B: rfft as GEMM  XF(b) = F(TPxP) x X(b)(PxC) ----------
// Epilogue repacks each 16x16 f32 tile through wave-private LDS so the
// (b,freq,c) output is written with 16-byte stores instead of 2-byte scatter.
__global__ __launch_bounds__(256) void gemm_rfft(
    const unsigned short* __restrict__ Fc, const unsigned short* __restrict__ Fs,
    const unsigned short* __restrict__ XcT,
    unsigned short* __restrict__ XFre, unsigned short* __restrict__ XFim) {
  __shared__ float lre[8][16][16];
  __shared__ float lim[8][16][16];
  int b = blockIdx.z;
  int wave = threadIdx.x >> 5;
  int tile = blockIdx.x * 8 + wave;                 // 256 tiles = 8 Mt x 32 Nt
  int mt = tile >> 5, nt = tile & 31;
  int m0 = mt * 16, n0 = nt * 16;
  const unsigned short* Bm = XcT + (size_t)b * kC * kP;
  v8f acc_re = {}, acc_im = {};
  for (int k0 = 0; k0 < kP; k0 += 32) {
    v16u bb = load_frag_bt(Bm, k0, n0, kP);
    v16u ac = load_frag_a(Fc, m0, k0, kP);
    v16u as = load_frag_a(Fs, m0, k0, kP);
    acc_re = wmma_bf16(ac, bb, acc_re);
    acc_im = wmma_bf16(as, bb, acc_im);
  }
  int lane = threadIdx.x & 31;
  int nrel = lane & 15;
  int mrel = (lane & 16) ? 8 : 0;
#pragma unroll
  for (int r = 0; r < 8; ++r) {
    lre[wave][mrel + r][nrel] = acc_re[r];
    lim[wave][mrel + r][nrel] = acc_im[r];
  }
  // Wave-private LDS, in-order DS ops within a wave: no barrier needed.
  int row  = lane & 15;
  int half = lane >> 4;                             // 8-column chunk
  v8u pkre, pkim;
#pragma unroll
  for (int j = 0; j < 8; ++j) {
    pkre[j] = f2bf(lre[wave][row][half * 8 + j]);
    pkim[j] = f2bf(lim[wave][row][half * 8 + j]);
  }
  size_t o = (size_t)b * kTP * kC + (size_t)(m0 + row) * kC + n0 + half * 8;
  *(v8u*)(XFre + o) = pkre;
  *(v8u*)(XFim + o) = pkim;
}

// ---------- Stage C: per-frequency complex linear (dominant GEMMs) ----------
// y[b,d] = sum_c xf[b,c] * W[kf][d,c]
// M=d (rows of W, natural layout), N=batch, K=c. Each wave computes BOTH
// batch N-tiles so every weight fragment is loaded exactly once per layer.
// yr = Wr*xr + Wi*(-xi); yi = Wi*xr + Wr*xi  (2 accumulators per N-tile).
__global__ __launch_bounds__(256) void gemm_cplx(
    const unsigned short* __restrict__ XFre, const unsigned short* __restrict__ XFim,
    const unsigned short* __restrict__ Wr_bf, const unsigned short* __restrict__ Wi_bf,
    const float* __restrict__ br, const float* __restrict__ bi,
    const float* __restrict__ wk,
    unsigned short* __restrict__ YrT, unsigned short* __restrict__ YiT) {
  int kf = blockIdx.y;                               // 0..T-1
  int m0 = (blockIdx.x * 8 + (threadIdx.x >> 5)) * 16;  // 32 Mt(d) tiles
  const unsigned short* Awr = Wr_bf + (size_t)kf * kC * kC;
  const unsigned short* Awi = Wi_bf + (size_t)kf * kC * kC;
  const unsigned short* Bre = XFre + (size_t)kf * kC;
  const unsigned short* Bim = XFim + (size_t)kf * kC;
  const size_t ldb = (size_t)kTP * kC;
  int lane = threadIdx.x & 31;
  v8f yr0 = {}, yi0 = {}, yr1 = {}, yi1 = {};
  for (int k0 = 0; k0 < kC; k0 += 32) {
    v16u wr_ = load_frag_a(Awr, m0, k0, kC);
    v16u wi_ = load_frag_a(Awi, m0, k0, kC);
    v16u xr0 = load_frag_bt(Bre, k0, 0, ldb);
    v16u xi0 = load_frag_bt(Bim, k0, 0, ldb);
    v16u xr1 = load_frag_bt(Bre, k0, 16, ldb);
    v16u xi1 = load_frag_bt(Bim, k0, 16, ldb);
    v16u xn0 = negbf(xi0);
    v16u xn1 = negbf(xi1);
    yr0 = wmma_bf16(wr_, xr0, yr0);
    yr0 = wmma_bf16(wi_, xn0, yr0);
    yi0 = wmma_bf16(wi_, xr0, yi0);
    yi0 = wmma_bf16(wr_, xi0, yi0);
    yr1 = wmma_bf16(wr_, xr1, yr1);
    yr1 = wmma_bf16(wi_, xn1, yr1);
    yi1 = wmma_bf16(wi_, xr1, yi1);
    yi1 = wmma_bf16(wr_, xi1, yi1);
    if (k0 + 32 < kC) {                 // stream next weight rows through L2
      const unsigned short* pf0 = Awr + (size_t)(m0 + (lane & 15)) * kC + (k0 + 32);
      const unsigned short* pf1 = Awi + (size_t)(m0 + (lane & 15)) * kC + (k0 + 32);
      __builtin_prefetch(pf0, 0, 1);
      __builtin_prefetch(pf1, 0, 1);
    }
  }
  int nrel = lane & 15;
  int mb = m0 + ((lane & 16) ? 8 : 0);               // d base
  float wkv = wk[kf];
  float brv[8], biv[8];
#pragma unroll
  for (int r = 0; r < 8; ++r) {
    brv[r] = br[(size_t)kf * kC + mb + r];
    biv[r] = bi[(size_t)kf * kC + mb + r];
  }
#pragma unroll
  for (int half = 0; half < 2; ++half) {
    v8f ayr = half ? yr1 : yr0;
    v8f ayi = half ? yi1 : yi0;
    int n = half * 16 + nrel;                        // batch index
#pragma unroll
    for (int r = 0; r < 8; ++r) {
      int d = mb + r;
      float yr = ayr[r] + brv[r];
      float yi = ayi[r] + biv[r];
      size_t o = ((size_t)n * kC + d) * kTP + kf;    // (b, c, freq) layout
      YrT[o] = f2bf(yr * wkv);
      YiT[o] = f2bf(yi * wkv);
    }
  }
}

// ---------- Stage D: inverse basis + residual ----------
__global__ __launch_bounds__(256) void gemm_inv(
    const unsigned short* __restrict__ Cb, const unsigned short* __restrict__ Sb,
    const unsigned short* __restrict__ YrT, const unsigned short* __restrict__ YiT,
    float* __restrict__ Xc, unsigned short* __restrict__ XcT) {
  int b = blockIdx.z;
  int tile = blockIdx.x * 8 + (threadIdx.x >> 5);    // 384 tiles = 12 Mt x 32 Nt
  int mt = tile >> 5, nt = tile & 31;
  int m0 = mt * 16, n0 = nt * 16;
  const unsigned short* Br = YrT + (size_t)b * kC * kTP;
  const unsigned short* Bi = YiT + (size_t)b * kC * kTP;
  v8f acc = {};
  for (int k0 = 0; k0 < kTP; k0 += 32) {
    v16u ar  = load_frag_a(Cb, m0, k0, kTP);
    v16u ai  = load_frag_a(Sb, m0, k0, kTP);
    v16u br_ = load_frag_bt(Br, k0, n0, kTP);
    v16u bi_ = load_frag_bt(Bi, k0, n0, kTP);
    acc = wmma_bf16(ar, br_, acc);
    acc = wmma_bf16(ai, bi_, acc);
  }
  int lane = threadIdx.x & 31;
  int n = n0 + (lane & 15);
  int mb = m0 + ((lane & 16) ? 8 : 0);
  float* x = Xc + (size_t)b * kP * kC;
  v8u pk;
#pragma unroll
  for (int r = 0; r < 8; ++r) {
    size_t o = (size_t)(mb + r) * kC + n;
    float v = x[o] + acc[r];                         // residual
    x[o] = v;
    pk[r] = f2bf(v);
  }
  *(v8u*)(XcT + ((size_t)b * kC + n) * kP + mb) = pk;
}

// ---------- Stage E: projection + de-normalize ----------
__global__ __launch_bounds__(256) void gemm_proj(
    const unsigned short* __restrict__ Wq_bf, const unsigned short* __restrict__ XcT,
    const float* __restrict__ bq, const float* __restrict__ stdev,
    const float* __restrict__ mean, float* __restrict__ out) {
  int b = blockIdx.z;
  int tile = blockIdx.x * 8 + (threadIdx.x >> 5);    // 384 tiles
  int mt = tile >> 5, nt = tile & 31;
  int m0 = mt * 16, n0 = nt * 16;
  const unsigned short* Bm = XcT + (size_t)b * kC * kP;
  v8f acc = {};
  for (int k0 = 0; k0 < kP; k0 += 32) {
    v16u a  = load_frag_a(Wq_bf, m0, k0, kP);
    v16u bb = load_frag_bt(Bm, k0, n0, kP);
    acc = wmma_bf16(a, bb, acc);
  }
  int lane = threadIdx.x & 31;
  int n = n0 + (lane & 15);
  int mb = m0 + ((lane & 16) ? 8 : 0);
  float sd = stdev[b * kC + n];
  float mn = mean[b * kC + n];
  float* o = out + (size_t)b * kP * kC;
#pragma unroll
  for (int r = 0; r < 8; ++r) {
    float v = acc[r] + bq[mb + r];
    o[(size_t)(mb + r) * kC + n] = v * sd + mn;
  }
}

extern "C" void kernel_launch(void* const* d_in, const int* in_sizes, int n_in,
                              void* d_out, int out_size, void* d_ws, size_t ws_size,
                              hipStream_t stream) {
  (void)in_sizes; (void)n_in; (void)out_size; (void)ws_size;
  const float* x_enc = (const float*)d_in[0];
  const float* Wp    = (const float*)d_in[4];
  const float* bp    = (const float*)d_in[5];
  const float* Wq    = (const float*)d_in[6];
  const float* bq    = (const float*)d_in[7];
  const float* Wr    = (const float*)d_in[8];
  const float* Wi    = (const float*)d_in[9];
  const float* br    = (const float*)d_in[10];
  const float* bi    = (const float*)d_in[11];
  const float* w     = (const float*)d_in[12];

  char* ws = (char*)d_ws;
  size_t off = 0;
  auto take = [&](size_t bytes) -> char* {
    char* p = ws + off;
    off += (bytes + 255) & ~(size_t)255;
    return p;
  };
  float* mean   = (float*)take((size_t)kB * kC * 4);
  float* stdev  = (float*)take((size_t)kB * kC * 4);
  float* invstd = (float*)take((size_t)kB * kC * 4);
  float* wk1    = (float*)take(kT * 4);
  float* wk2    = (float*)take(kT * 4);
  unsigned short* Wp_bf = (unsigned short*)take((size_t)kP * kL * 2);
  unsigned short* Wq_bf = (unsigned short*)take((size_t)kP * kP * 2);
  unsigned short* Fc    = (unsigned short*)take((size_t)kTP * kP * 2);
  unsigned short* Fs    = (unsigned short*)take((size_t)kTP * kP * 2);
  unsigned short* Cb    = (unsigned short*)take((size_t)kP * kTP * 2);
  unsigned short* Sb    = (unsigned short*)take((size_t)kP * kTP * 2);
  unsigned short* Wr_bf = (unsigned short*)take((size_t)kT * kC * kC * 2);
  unsigned short* Wi_bf = (unsigned short*)take((size_t)kT * kC * kC * 2);
  // XnT (B*C*L bf16, 16 MB) is dead after stage A; the four B*TP*C bf16 buffers
  // (4 MB each, exactly 16 MB total) alias the same region.
  char* xn_block = take((size_t)kB * kC * kL * 2);
  unsigned short* XnT  = (unsigned short*)xn_block;
  unsigned short* XFre = (unsigned short*)xn_block;
  unsigned short* XFim = XFre + (size_t)kB * kTP * kC;
  unsigned short* YrT  = XFim + (size_t)kB * kTP * kC;
  unsigned short* YiT  = YrT  + (size_t)kB * kTP * kC;
  float*          Xc   = (float*)take((size_t)kB * kP * kC * 4);
  unsigned short* XcT  = (unsigned short*)take((size_t)kB * kC * kP * 2);

  // --- preprocessing ---
  stats_kernel<<<kB, 256, 0, stream>>>(x_enc, mean, stdev, invstd);
  normalize_t_kernel<<<dim3(kC / 32, kL / 32, kB), 256, 0, stream>>>(x_enc, mean, invstd, XnT);
  cvt_bf16_kernel<<<96, 256, 0, stream>>>(Wp, Wp_bf, (size_t)kP * kL / 4);
  cvt_bf16_kernel<<<36, 256, 0, stream>>>(Wq, Wq_bf, (size_t)kP * kP / 4);
  cvt_bf16_kernel<<<8192, 256, 0, stream>>>(Wr, Wr_bf, (size_t)kT * kC * kC / 4);
  cvt_bf16_kernel<<<8192, 256, 0, stream>>>(Wi, Wi_bf, (size_t)kT * kC * kC / 4);
  basis_kernel<<<(kTP * kP + 255) / 256, 256, 0, stream>>>(Fc, Fs, Cb, Sb);
  wk_kernel<<<1, 128, 0, stream>>>(w, wk1, wk2);

  // --- stage A: predict_linear (consumes XnT) ---
  gemm_predict<<<dim3(48, 1, kB), 256, 0, stream>>>(Wp_bf, XnT, bp, Xc, XcT);

  // Zero YrT/YiT (padded freq rows T..TP-1 must be zero). Safe now: XnT is dead.
  hipMemsetAsync(YrT, 0, (size_t)2 * kB * kTP * kC * 2, stream);

  // --- two residual FFT-decomposition layers ---
  for (int layer = 0; layer < 2; ++layer) {
    gemm_rfft<<<dim3(32, 1, kB), 256, 0, stream>>>(Fc, Fs, XcT, XFre, XFim);
    gemm_cplx<<<dim3(4, kT), 256, 0, stream>>>(XFre, XFim, Wr_bf, Wi_bf, br, bi,
                                               layer == 0 ? wk1 : wk2, YrT, YiT);
    gemm_inv<<<dim3(48, 1, kB), 256, 0, stream>>>(Cb, Sb, YrT, YiT, Xc, XcT);
  }

  // --- stage E: projection + de-normalize ---
  gemm_proj<<<dim3(48, 1, kB), 256, 0, stream>>>(Wq_bf, XcT, bq, stdev, mean,
                                                 (float*)d_out);
}
